// MultiHeadAttention_64450279244456
// MI455X (gfx1250) — matrix operand order repaired and checked by reading, not compiled, not run on hardware
//
#include <hip/hip_runtime.h>
#include <hip/hip_bf16.h>

// MI455X (gfx1250) fused multi-head attention block.
// f16 WMMA (v_wmma_f32_16x16x32_f16) for all GEMMs, f32 accumulation.
// Intermediates q/k/v/context kept in f16 in workspace; v stored transposed
// per head so the attn@v B-operand is contiguous. 32-row M-tiles in the big
// GEMMs so each B (weight) fragment feeds two WMMAs.

typedef __attribute__((ext_vector_type(16))) _Float16 v16h;
typedef __attribute__((ext_vector_type(8)))  _Float16 v8h;
typedef __attribute__((ext_vector_type(8)))  float    v8f;

#define BATCH 1024
#define SEQ   50
#define DM    512
#define NH    8
#define DK    64

__device__ __forceinline__ v8f wmma_f16(v16h a, v16h b, v8f c) {
    return __builtin_amdgcn_wmma_f32_16x16x32_f16(
        /*neg_a=*/false, a, /*neg_b=*/false, b,
        /*c_mod=*/(short)0, c, /*reuse_a=*/false, /*reuse_b=*/false);
}

// A fragment (16x32, f16): lane holds row (lane&15); lanes<16 take K={0..7,16..23},
// lanes>=16 take K={8..15,24..31}. p0 = base + row*ldh + k0 (halves).
__device__ __forceinline__ v16h frag_a(const _Float16* p0, int lane) {
    int hs = (lane >> 4) ? 8 : 0;
    v16h a;
    ((v8h*)&a)[0] = *(const v8h*)(p0 + hs);
    ((v8h*)&a)[1] = *(const v8h*)(p0 + hs + 16);
    return a;
}

// Same A fragment but sourced from f32 memory with in-register convert.
__device__ __forceinline__ v16h frag_a_f32(const float* p0, int lane) {
    int hs = (lane >> 4) ? 8 : 0;
    v16h a;
#pragma unroll
    for (int j = 0; j < 8; ++j) {
        a[j]     = (_Float16)p0[hs + j];
        a[8 + j] = (_Float16)p0[hs + 16 + j];
    }
    return a;
}

// B fragment (32x16, f16): lane holds column (lane&15); lanes<16 take K={0..7,16..23},
// lanes>=16 take K={8..15,24..31} (matches A). p0 = baseT + col*ldh + k0.
__device__ __forceinline__ v16h frag_b(const _Float16* p0, int lane) {
    int hs = (lane >> 4) ? 8 : 0;
    v16h b;
    ((v8h*)&b)[0] = *(const v8h*)(p0 + hs);
    ((v8h*)&b)[1] = *(const v8h*)(p0 + hs + 16);
    return b;
}

// ---------------------------------------------------------------------------
// Kernel 1: convert the four 512x512 f32 weights to f16 (order WQ,WK,WV,WFC).
__global__ void k_cvt_w(const float* __restrict__ wq, const float* __restrict__ wk,
                        const float* __restrict__ wv, const float* __restrict__ wfc,
                        _Float16* __restrict__ out) {
    int r = blockIdx.x * blockDim.x + threadIdx.x;       // 0 .. DM*DM-1
    int w = blockIdx.y;                                  // 0..3
    const float* src = (w == 0) ? wq : (w == 1) ? wk : (w == 2) ? wv : wfc;
    out[(size_t)w * DM * DM + r] = (_Float16)src[r];
}

// ---------------------------------------------------------------------------
// Kernel 2: Y = X @ W^T for Q/K/V (blockIdx.y selects projection).
// Block = 256 threads (8 waves); block owns 32 rows x all 512 cols.
// Each wave: 32(M) x 64(N) tile -> every B fragment feeds two WMMAs.
// q,k stored [B,H,S,64] f16; v stored transposed+padded [B,H,64,64] f16.
__global__ __launch_bounds__(256)
void k_proj(const float* __restrict__ Qi, const float* __restrict__ Ki,
            const float* __restrict__ Vi, const _Float16* __restrict__ Wh,
            _Float16* __restrict__ qh, _Float16* __restrict__ kh,
            _Float16* __restrict__ vT) {
    const int proj = blockIdx.y;
    const float* X = (proj == 0) ? Qi : (proj == 1) ? Ki : Vi;
    const _Float16* W = Wh + (size_t)proj * DM * DM;

    const int lane  = threadIdx.x & 31;
    const int wave  = threadIdx.x >> 5;       // 0..7 -> N strip
    const int ln    = lane & 15;
    const int hi    = lane >> 4;
    const int mBase = blockIdx.x * 32;
    const int nBase = wave * 64;

    v8f c[2][4] = {};
    for (int k0 = 0; k0 < DM; k0 += 32) {
        v16h a0 = frag_a_f32(X + (size_t)(mBase + ln) * DM + k0, lane);
        v16h a1 = frag_a_f32(X + (size_t)(mBase + 16 + ln) * DM + k0, lane);
#pragma unroll
        for (int t = 0; t < 4; ++t) {
            v16h b = frag_b(W + (size_t)(nBase + t * 16 + ln) * DM + k0, lane);
            c[0][t] = wmma_f16(a0, b, c[0][t]);
            c[1][t] = wmma_f16(a1, b, c[1][t]);
        }
    }
#pragma unroll
    for (int u = 0; u < 2; ++u) {
#pragma unroll
        for (int i = 0; i < 8; ++i) {
            int m  = mBase + u * 16 + i + 8 * hi;
            int b_ = m / SEQ, s = m - b_ * SEQ;          // hoisted per row
#pragma unroll
            for (int t = 0; t < 4; ++t) {
                int n = nBase + t * 16 + ln;
                int h = n >> 6, dk = n & 63;
                _Float16 val = (_Float16)c[u][t][i];
                if (proj == 0)
                    qh[((size_t)(b_ * NH + h) * SEQ + s) * DK + dk] = val;
                else if (proj == 1)
                    kh[((size_t)(b_ * NH + h) * SEQ + s) * DK + dk] = val;
                else
                    vT[((size_t)(b_ * NH + h) * DK + dk) * 64 + s] = val;
            }
        }
    }
}

// ---------------------------------------------------------------------------
// Kernel 3: attention for one (batch, head) per block; 4 waves x 16-query strips.
__global__ __launch_bounds__(128)
void k_attn(const _Float16* __restrict__ qh, const _Float16* __restrict__ kh,
            const _Float16* __restrict__ vT, const unsigned char* __restrict__ msk,
            const float* __restrict__ tw, _Float16* __restrict__ ctx) {
    const int bh = blockIdx.x;              // 0..B*NH-1
    const int b  = bh >> 3, h = bh & 7;
    const int lane  = threadIdx.x & 31;
    const int wave  = threadIdx.x >> 5;     // 0..3
    const int ln    = lane & 15;
    const int hi    = lane >> 4;
    const int sBase = wave * 16;

    const _Float16* qp = qh + (size_t)bh * SEQ * DK;
    const _Float16* kp = kh + (size_t)bh * SEQ * DK;
    const _Float16* vp = vT + (size_t)bh * DK * 64;

    __shared__ _Float16 s_attn[4][16][64];

    int qrow = sBase + ln; if (qrow > SEQ - 1) qrow = SEQ - 1;

    // scores = q @ k^T  (M=16 strip, N=64 padded keys, K=64)
    v8f sc[4] = {};
    for (int k0 = 0; k0 < DK; k0 += 32) {
        v16h a = frag_a(qp + qrow * DK + k0, lane);
#pragma unroll
        for (int t = 0; t < 4; ++t) {
            int key = t * 16 + ln;
            int kc  = (key > SEQ - 1) ? SEQ - 1 : key;
            v16h bf = frag_b(kp + kc * DK + k0, lane);
            sc[t] = wmma_f16(a, bf, sc[t]);
        }
    }

    // scale + mask + softmax + time_weight, write f16 attn tile to LDS
#pragma unroll
    for (int i = 0; i < 8; ++i) {
        int rrow = sBase + i + 8 * hi;
        int rc   = (rrow > SEQ - 1) ? SEQ - 1 : rrow;
        float pv[4];
        float mx = -3.0e38f;
#pragma unroll
        for (int t = 0; t < 4; ++t) {
            int key = t * 16 + ln;
            float v = sc[t][i] * 0.125f;                 // 1/sqrt(64)
            if (key >= SEQ) v = -1.0e30f;                // pad columns -> exp==0
            else if (msk[((size_t)b * SEQ + rc) * SEQ + key]) v = -1.0e9f;
            pv[t] = v;
            mx = fmaxf(mx, v);
        }
#pragma unroll
        for (int d = 1; d < 16; d <<= 1) mx = fmaxf(mx, __shfl_xor(mx, d, 32));
        float sum = 0.f;
#pragma unroll
        for (int t = 0; t < 4; ++t) { float e = __expf(pv[t] - mx); pv[t] = e; sum += e; }
#pragma unroll
        for (int d = 1; d < 16; d <<= 1) sum += __shfl_xor(sum, d, 32);
        float inv = 1.0f / (sum + 1e-20f);
#pragma unroll
        for (int t = 0; t < 4; ++t) {
            int key = t * 16 + ln;
            float e = pv[t] * inv;
            if (key < SEQ) e *= tw[((size_t)h * SEQ + rc) * SEQ + key];
            s_attn[wave][i + 8 * hi][t * 16 + ln] = (_Float16)e;
        }
    }
    __syncthreads();

    // context = attn @ v  (M=16 strip, N=64 dk, K=64 padded keys)
    v8f cc[4] = {};
    const _Float16* ap = &s_attn[wave][0][0];
    for (int k0 = 0; k0 < 64; k0 += 32) {
        v16h a = frag_a(ap + ln * 64 + k0, lane);
#pragma unroll
        for (int t = 0; t < 4; ++t) {
            int dk = t * 16 + ln;
            v16h bf = frag_b(vp + dk * 64 + k0, lane);
            cc[t] = wmma_f16(a, bf, cc[t]);
        }
    }
#pragma unroll
    for (int i = 0; i < 8; ++i) {
        int s = sBase + i + 8 * hi;
        if (s < SEQ) {
            size_t rb = ((size_t)b * SEQ + s) * DM + h * DK;
#pragma unroll
            for (int t = 0; t < 4; ++t)
                ctx[rb + t * 16 + ln] = (_Float16)cc[t][i];
        }
    }
}

// ---------------------------------------------------------------------------
// Kernel 4: out = LayerNorm(ctx @ Wfc^T + residual). Block owns 32 full rows.
__global__ __launch_bounds__(256)
void k_out(const _Float16* __restrict__ ctx, const _Float16* __restrict__ Wfc,
           const float* __restrict__ Qres, float* __restrict__ out) {
    const int lane  = threadIdx.x & 31;
    const int wave  = threadIdx.x >> 5;     // 0..7 -> N strip
    const int ln    = lane & 15;
    const int hi    = lane >> 4;
    const int mBase = blockIdx.x * 32;
    const int nBase = wave * 64;

    __shared__ float s_row[32][DM];          // 64 KB
    __shared__ float s_mu[32], s_rs[32];

    v8f c[2][4] = {};
    for (int k0 = 0; k0 < DM; k0 += 32) {
        v16h a0 = frag_a(ctx + (size_t)(mBase + ln) * DM + k0, lane);
        v16h a1 = frag_a(ctx + (size_t)(mBase + 16 + ln) * DM + k0, lane);
#pragma unroll
        for (int t = 0; t < 4; ++t) {
            v16h b = frag_b(Wfc + (size_t)(nBase + t * 16 + ln) * DM + k0, lane);
            c[0][t] = wmma_f16(a0, b, c[0][t]);
            c[1][t] = wmma_f16(a1, b, c[1][t]);
        }
    }
#pragma unroll
    for (int u = 0; u < 2; ++u) {
#pragma unroll
        for (int t = 0; t < 4; ++t) {
            int n = nBase + t * 16 + ln;
#pragma unroll
            for (int i = 0; i < 8; ++i) {
                int r = u * 16 + i + 8 * hi;
                s_row[r][n] = c[u][t][i] + Qres[(size_t)(mBase + r) * DM + n];
            }
        }
    }
    __syncthreads();

    // LayerNorm stats: 32 rows / 8 waves -> 4 rows per wave (16 lanes per row)
#pragma unroll
    for (int rr = 0; rr < 2; ++rr) {
        int r = wave * 4 + rr * 2 + hi;
        float s = 0.f;
#pragma unroll
        for (int j = 0; j < DM / 16; ++j) s += s_row[r][ln + j * 16];
#pragma unroll
        for (int d = 1; d < 16; d <<= 1) s += __shfl_xor(s, d, 32);
        float mu = s * (1.0f / DM);
        float vs = 0.f;
#pragma unroll
        for (int j = 0; j < DM / 16; ++j) {
            float x = s_row[r][ln + j * 16] - mu;
            vs += x * x;
        }
#pragma unroll
        for (int d = 1; d < 16; d <<= 1) vs += __shfl_xor(vs, d, 32);
        if (ln == 0) { s_mu[r] = mu; s_rs[r] = rsqrtf(vs * (1.0f / DM) + 1e-5f); }
    }
    __syncthreads();

    const int tid = threadIdx.x;
#pragma unroll
    for (int k = 0; k < 64; ++k) {           // 32*512 / 256 threads
        int idx = tid + k * 256;
        int i = idx >> 9, n = idx & 511;
        out[(size_t)(mBase + i) * DM + n] = (s_row[i][n] - s_mu[i]) * s_rs[i];
    }
}

// ---------------------------------------------------------------------------
extern "C" void kernel_launch(void* const* d_in, const int* in_sizes, int n_in,
                              void* d_out, int out_size, void* d_ws, size_t ws_size,
                              hipStream_t stream) {
    const float* Q   = (const float*)d_in[0];
    const float* K   = (const float*)d_in[1];
    const float* V   = (const float*)d_in[2];
    const unsigned char* msk = (const unsigned char*)d_in[3];
    const float* WQ  = (const float*)d_in[4];
    const float* WK  = (const float*)d_in[5];
    const float* WV  = (const float*)d_in[6];
    const float* WFC = (const float*)d_in[7];
    const float* TW  = (const float*)d_in[8];
    float* out = (float*)d_out;

    char* ws = (char*)d_ws;
    _Float16* Wh  = (_Float16*)ws;  ws += (size_t)4 * DM * DM * 2;           // 2 MB
    _Float16* qh  = (_Float16*)ws;  ws += (size_t)BATCH * NH * SEQ * DK * 2; // 52 MB
    _Float16* kh  = (_Float16*)ws;  ws += (size_t)BATCH * NH * SEQ * DK * 2; // 52 MB
    _Float16* vT  = (_Float16*)ws;  ws += (size_t)BATCH * NH * DK * 64 * 2;  // 64 MB
    _Float16* ctx = (_Float16*)ws;  ws += (size_t)BATCH * SEQ * DM * 2;      // 52 MB

    k_cvt_w<<<dim3(DM * DM / 256, 4), 256, 0, stream>>>(WQ, WK, WV, WFC, Wh);
    k_proj<<<dim3(BATCH * SEQ / 32, 3), 256, 0, stream>>>(Q, K, V, Wh, qh, kh, vT);
    k_attn<<<dim3(BATCH * NH), 128, 0, stream>>>(qh, kh, vT, msk, TW, ctx);
    k_out<<<dim3(BATCH * SEQ / 32), 256, 0, stream>>>(ctx, Wh + (size_t)3 * DM * DM, Q, out);
}
